// GATLayer_55808805044407
// MI455X (gfx1250) — matrix operand order repaired and checked by reading, not compiled, not run on hardware
//
#include <hip/hip_runtime.h>

// ---------------- types ----------------
typedef __attribute__((ext_vector_type(16))) __bf16 v16bf;
typedef __attribute__((ext_vector_type(8)))  float  v8f;

#define N_NODES 8192
#define F_INK   512
#define F_OUTK  256
#define ALPHA   0.2f
#define LOG2E   1.4426950408889634f

// LDS row pitch for staged hT chunk: 32 bf16 (64B) padded to 80B -> conflict-free b128 reads
#define ROWU    40   // ushorts per row (80 bytes)

union BF16x16 { v16bf v; unsigned short s[16]; uint4 u[2]; };
union BF16x8  { uint4 u; unsigned short s[8]; };

__device__ __forceinline__ unsigned short f2bf(float f) {
    __bf16 b = (__bf16)f;                    // RNE convert, v_cvt_pk_bf16_f32
    union { __bf16 b; unsigned short u; } c; c.b = b; return c.u;
}
__device__ __forceinline__ float bf2f(unsigned short u) {
    union { unsigned int i; float f; } c; c.i = ((unsigned int)u) << 16; return c.f;
}

// gfx1250 async global->LDS copy (ASYNCcnt); 16B per lane
__device__ __forceinline__ void async_load_b128(unsigned lds_off, const void* gaddr) {
    asm volatile("global_load_async_to_lds_b128 %0, %1, off"
                 :: "v"(lds_off), "v"(gaddr)
                 : "memory");
}
__device__ __forceinline__ void wait_async0() {
    asm volatile("s_wait_asynccnt 0" ::: "memory");
}

// ---------------- kernel 1: W (512x256 f32) -> WT (256x512 bf16) ----------------
__global__ __launch_bounds__(256) void wt_kernel(const float* __restrict__ W,
                                                 unsigned short* __restrict__ WT) {
    int idx = blockIdx.x * blockDim.x + threadIdx.x;
    int k = idx >> 8;
    int n = idx & 255;
    WT[n * F_INK + k] = f2bf(W[idx]);
}

// ---------------- kernel 2: hT (256 x 8192 bf16) = (X @ W)^T via WMMA bf16 ----------------
__global__ __launch_bounds__(128) void h_gemm_kernel(const float* __restrict__ X,
                                                     const unsigned short* __restrict__ WT,
                                                     unsigned short* __restrict__ hT) {
    int wave = blockIdx.x * 4 + (threadIdx.x >> 5);
    int lane = threadIdx.x & 31;
    int r    = lane & 15;
    int g    = lane >> 4;
    int mTile = wave >> 4;
    int nTile = wave & 15;
    int i0 = mTile * 16, n0 = nTile * 16;

    v8f acc = { 0.f, 0.f, 0.f, 0.f, 0.f, 0.f, 0.f, 0.f };

    const float*          arow = X  + (size_t)(i0 + r) * F_INK + g * 8;
    const unsigned short* brow = WT + (size_t)(n0 + r) * F_INK + g * 8;

    for (int k0 = 0; k0 < F_INK; k0 += 32) {
        float4 f0 = *(const float4*)(arow + k0);
        float4 f1 = *(const float4*)(arow + k0 + 4);
        float4 f2 = *(const float4*)(arow + k0 + 16);
        float4 f3 = *(const float4*)(arow + k0 + 20);
        BF16x16 A;
        A.s[0]=f2bf(f0.x); A.s[1]=f2bf(f0.y); A.s[2]=f2bf(f0.z); A.s[3]=f2bf(f0.w);
        A.s[4]=f2bf(f1.x); A.s[5]=f2bf(f1.y); A.s[6]=f2bf(f1.z); A.s[7]=f2bf(f1.w);
        A.s[8]=f2bf(f2.x); A.s[9]=f2bf(f2.y); A.s[10]=f2bf(f2.z); A.s[11]=f2bf(f2.w);
        A.s[12]=f2bf(f3.x); A.s[13]=f2bf(f3.y); A.s[14]=f2bf(f3.z); A.s[15]=f2bf(f3.w);
        BF16x16 B;
        B.u[0] = *(const uint4*)(brow + k0);
        B.u[1] = *(const uint4*)(brow + k0 + 16);
        acc = __builtin_amdgcn_wmma_f32_16x16x32_bf16(false, A.v, false, B.v,
                                                      (short)0, acc, false, false);
    }
    BF16x8 st;
    #pragma unroll
    for (int q = 0; q < 8; ++q) st.s[q] = f2bf(acc[q]);
    *(uint4*)(hT + (size_t)(n0 + r) * N_NODES + i0 + 8 * g) = st.u;
}

// ---------------- kernel 3: s1 = h @ a1, s2 = h @ a2 ----------------
__global__ __launch_bounds__(256) void sv_kernel(const unsigned short* __restrict__ hT,
                                                 const float* __restrict__ a,
                                                 float* __restrict__ s1,
                                                 float* __restrict__ s2) {
    int i = blockIdx.x * blockDim.x + threadIdx.x;
    float acc1 = 0.f, acc2 = 0.f;
    for (int f = 0; f < F_OUTK; ++f) {
        float hv = bf2f(hT[(size_t)f * N_NODES + i]);
        acc1 += hv * a[f];
        acc2 += hv * a[F_OUTK + f];
    }
    s1[i] = acc1;
    s2[i] = acc2;
}

// ---------------- kernel 4: S2MAX = max(s2) ----------------
__global__ __launch_bounds__(1024) void s2max_kernel(const float* __restrict__ s2,
                                                     float* __restrict__ s2max) {
    __shared__ float red[1024];
    int t = threadIdx.x;
    float m = -3.4e38f;
    #pragma unroll
    for (int k = 0; k < 8; ++k) m = fmaxf(m, s2[t + 1024 * k]);
    red[t] = m;
    __syncthreads();
    for (int s = 512; s > 0; s >>= 1) {
        if (t < s) red[t] = fmaxf(red[t], red[t + s]);
        __syncthreads();
    }
    if (t == 0) *s2max = red[0];
}

// ---------------- kernel 5: fused masked-softmax + attn @ h ----------------
// 4 waves/block, each wave owns a 16-row tile; block walks j chunks in lockstep.
// hT chunk (32 x 256 bf16) is double-buffered in LDS via async global->LDS copies.
__global__ __launch_bounds__(128) void attn_kernel(const int* __restrict__ adj,
                                                   const unsigned short* __restrict__ hT,
                                                   const float* __restrict__ s1,
                                                   const float* __restrict__ s2,
                                                   const float* __restrict__ s2maxp,
                                                   float* __restrict__ out) {
    // [buf][feature row f: 0..255][32 K values padded to 80B]
    __shared__ __align__(16) unsigned short sbuf[2][256 * ROWU];

    int tid  = threadIdx.x;
    int wave = blockIdx.x * 4 + (tid >> 5);
    int lane = tid & 31;
    int r    = lane & 15;
    int g    = lane >> 4;
    int i0   = wave * 16;

    float s1r   = s1[i0 + r];
    float s2max = *s2maxp;
    float mrow  = s1r + s2max;                 // upper bound of row max (lrelu is monotone)
    mrow = mrow > 0.f ? mrow : ALPHA * mrow;

    v8f acc[16];
    v8f zero = { 0.f, 0.f, 0.f, 0.f, 0.f, 0.f, 0.f, 0.f };
    #pragma unroll
    for (int t = 0; t < 16; ++t) acc[t] = zero;

    const int*   arow = adj + (size_t)(i0 + r) * N_NODES + g * 8;
    const float* s2p  = s2  + g * 8;
    float lsum = 0.f;

    // prologue: stage chunk 0 into buffer 0 (8 x 128 lanes x 16B = 16KB)
    #pragma unroll
    for (int it = 0; it < 8; ++it) {
        int li = it * 128 + tid;               // 0..1023
        int f  = li >> 2;                      // feature row
        int p  = li & 3;                       // 16B piece within the 64B row
        unsigned loff = (unsigned)(size_t)&sbuf[0][f * ROWU + p * 8];
        async_load_b128(loff, hT + (size_t)f * N_NODES + p * 8);
    }

    for (int n = 0; n < N_NODES / 32; ++n) {
        int j0  = n * 32;
        int cur = n & 1;

        wait_async0();          // my copies into sbuf[cur] done
        __syncthreads();        // everyone's copies done; everyone done reading sbuf[cur^1]

        if (n + 1 < N_NODES / 32) {            // stage next chunk into the other buffer
            int jn = j0 + 32;
            #pragma unroll
            for (int it = 0; it < 8; ++it) {
                int li = it * 128 + tid;
                int f  = li >> 2;
                int p  = li & 3;
                unsigned loff = (unsigned)(size_t)&sbuf[cur ^ 1][f * ROWU + p * 8];
                async_load_b128(loff, hT + (size_t)f * N_NODES + jn + p * 8);
            }
        }

        // ---- adj tile in A-fragment layout + attention probabilities ----
        int4 a0 = *(const int4*)(arow + j0);
        int4 a1 = *(const int4*)(arow + j0 + 4);
        int4 a2 = *(const int4*)(arow + j0 + 16);
        int4 a3 = *(const int4*)(arow + j0 + 20);
        float4 t0 = *(const float4*)(s2p + j0);
        float4 t1 = *(const float4*)(s2p + j0 + 4);
        float4 t2 = *(const float4*)(s2p + j0 + 16);
        float4 t3 = *(const float4*)(s2p + j0 + 20);

        int   av[16] = { a0.x,a0.y,a0.z,a0.w, a1.x,a1.y,a1.z,a1.w,
                         a2.x,a2.y,a2.z,a2.w, a3.x,a3.y,a3.z,a3.w };
        float sv[16] = { t0.x,t0.y,t0.z,t0.w, t1.x,t1.y,t1.z,t1.w,
                         t2.x,t2.y,t2.z,t2.w, t3.x,t3.y,t3.z,t3.w };

        BF16x16 P;
        #pragma unroll
        for (int s = 0; s < 16; ++s) {
            float e = s1r + sv[s];
            e = e > 0.f ? e : ALPHA * e;                       // leaky relu
            float p = (av[s] > 0) ? exp2f((e - mrow) * LOG2E)  // native v_exp_f32
                                  : 0.f;
            lsum += p;
            P.s[s] = f2bf(p);
        }

        // ---- 16 WMMAs; B fragments come from LDS (ds_load_b128 x2 per tile) ----
        const unsigned short* hb = &sbuf[cur][r * ROWU + g * 8];
        #pragma unroll
        for (int t = 0; t < 16; ++t) {
            const unsigned short* bp = hb + t * 16 * ROWU;
            BF16x16 B;
            B.u[0] = *(const uint4*)(bp);
            B.u[1] = *(const uint4*)(bp + 16);
            acc[t] = __builtin_amdgcn_wmma_f32_16x16x32_bf16(false, P.v, false, B.v,
                                                             (short)0, acc[t], false, false);
        }

        // keep the adj HBM stream ahead of us
        __builtin_prefetch(arow + j0 + 512, 0, 0);
    }

    // combine half-lane partial sums per row, divide, store
    lsum += __shfl_xor(lsum, 16, 32);
    float inv = lsum > 0.f ? 1.f / lsum : 0.f;
    float invq[8];
    #pragma unroll
    for (int q = 0; q < 8; ++q) invq[q] = __shfl(inv, 8 * g + q, 32);

    #pragma unroll
    for (int t = 0; t < 16; ++t) {
        #pragma unroll
        for (int q = 0; q < 8; ++q) {
            out[(size_t)(i0 + 8 * g + q) * F_OUTK + t * 16 + r] = acc[t][q] * invq[q];
        }
    }
}

// ---------------- launcher ----------------
extern "C" void kernel_launch(void* const* d_in, const int* in_sizes, int n_in,
                              void* d_out, int out_size, void* d_ws, size_t ws_size,
                              hipStream_t stream) {
    const float* X   = (const float*)d_in[0];   // [8192,512]
    const float* W   = (const float*)d_in[1];   // [512,256]
    const float* a   = (const float*)d_in[2];   // [512,1]
    const int*   adj = (const int*)d_in[3];     // [8192,8192]
    float*       out = (float*)d_out;           // [8192,256]

    char* ws = (char*)d_ws;
    unsigned short* WT    = (unsigned short*)ws;                              // 256 KB
    unsigned short* hT    = (unsigned short*)(ws + (256u << 10));             // 4 MB
    float*          s1    = (float*)(ws + (256u << 10) + (4u << 20));         // 32 KB
    float*          s2    = s1 + N_NODES;                                     // 32 KB
    float*          s2max = s2 + N_NODES;                                     // 4 B

    wt_kernel    <<<(F_INK * F_OUTK) / 256, 256, 0, stream>>>(W, WT);
    h_gemm_kernel<<<(N_NODES / 16) * (F_OUTK / 16) / 4, 128, 0, stream>>>(X, WT, hT);
    sv_kernel    <<<N_NODES / 256, 256, 0, stream>>>(hT, a, s1, s2);
    s2max_kernel <<<1, 1024, 0, stream>>>(s2, s2max);
    attn_kernel  <<<(N_NODES / 16) / 4, 128, 0, stream>>>(adj, hT, s1, s2, s2max, out);
}
